// QuantumGate_65481071402619
// MI455X (gfx1250) — matrix operand order repaired and checked by reading, not compiled, not run on hardware
//
#include <hip/hip_runtime.h>
#include <math.h>

typedef _Float16 h2   __attribute__((ext_vector_type(2)));
typedef _Float16 v16h __attribute__((ext_vector_type(16)));
typedef float    v8f  __attribute__((ext_vector_type(8)));

#define NQ     16
#define QDIM   65536      // 2^16 amplitudes
#define QDEPTH 4
#define BLK    256        // 8 waves of 32

// ---------------------------------------------------------------------------
// Compile-time CNOT permutation algebra.
// Each layer applies CNOT(q,q+1) for q=0..14 then CNOT(15,0): a GF(2)-linear
// map F on the 16 index bits (qubit q <-> bit 15-q).  Instead of moving data,
// we track sigma_d : logical index -> physical LDS index (also GF(2)-linear),
// with sigma_0 = id and sigma_{d+1} = sigma_d o F^{-1}.  All masks are
// compile-time constants.
// ---------------------------------------------------------------------------
struct PermTables { unsigned S[QDEPTH + 1][16]; };

constexpr PermTables make_tables() {
    // Forward map F: w[p] = mask of original bits feeding output bit p.
    unsigned w[16];
    for (int p = 0; p < 16; ++p) w[p] = 1u << p;
    for (int P = 15; P >= 1; --P) w[P - 1] ^= w[P];  // CNOT(q,q+1), q=0..14
    w[15] ^= w[0];                                    // CNOT(15,0)
    // Invert F over GF(2) (Gauss-Jordan, rows as masks).
    unsigned A[16], Inv[16];
    for (int p = 0; p < 16; ++p) { A[p] = w[p]; Inv[p] = 1u << p; }
    for (int col = 0; col < 16; ++col) {
        int piv = col;
        while (piv < 16 && !((A[piv] >> col) & 1)) ++piv;
        if (piv >= 16) piv = col;                     // unreachable: F invertible
        unsigned ta = A[col]; A[col] = A[piv]; A[piv] = ta;
        unsigned ti = Inv[col]; Inv[col] = Inv[piv]; Inv[piv] = ti;
        for (int r = 0; r < 16; ++r)
            if (r != col && ((A[r] >> col) & 1)) { A[r] ^= A[col]; Inv[r] ^= Inv[col]; }
    }
    // Column masks of G=F^{-1}:  Gcol[p] = { j : p in G_j }  (== G applied to e_p)
    unsigned Gcol[16];
    for (int p = 0; p < 16; ++p) {
        unsigned cm = 0;
        for (int j = 0; j < 16; ++j) if ((Inv[j] >> p) & 1) cm |= 1u << j;
        Gcol[p] = cm;
    }
    // sigma tables: S[d][p] = physical mask of logical bit p during layer d.
    PermTables t{};
    for (int p = 0; p < 16; ++p) t.S[0][p] = 1u << p;
    for (int d = 0; d < QDEPTH; ++d)
        for (int p = 0; p < 16; ++p) {
            unsigned v = 0;
            for (int j = 0; j < 16; ++j)
                if ((Gcol[p] >> j) & 1) v ^= t.S[d][j];
            t.S[d + 1][p] = v;
        }
    return t;
}
static constexpr PermTables kPerm = make_tables();

static __device__ inline h2 mk_h2(float re, float im) {
    h2 r; r.x = (_Float16)re; r.y = (_Float16)im; return r;
}

// One workgroup simulates one batch element entirely in LDS.
// State: QDIM complex amplitudes stored as h2 (f16 re, f16 im) = 256 KB.
__global__ __launch_bounds__(BLK, 1)
void qsim_kernel(const float* __restrict__ x,      // [B,16]
                 const float* __restrict__ params, // [4,16,3]
                 float* __restrict__ out)          // [B,16]
{
    extern __shared__ unsigned char smem[];
    h2*       st    = (h2*)smem;                              // 256 KB state
    float*    gates = (float*)(smem + (size_t)QDIM * 4);      // [4][16][4]{re,im} = 2 KB
    _Float16* Are   = (_Float16*)(gates + QDEPTH * NQ * 8);   // 16x32 f16 = 1 KB
    _Float16* Aim   = Are + 512;                              // 16x32 f16 = 1 KB
    float*    red   = (float*)(Aim + 512);                    // 16 f32

    const int b    = blockIdx.x;
    const int tid  = threadIdx.x;
    const int lane = tid & 31;
    const int wave = tid >> 5;

    // ---- init state |0...0> (sigma(0)==0 always) ----
    for (int i = tid; i < QDIM; i += BLK) st[i] = mk_h2(0.f, 0.f);
    if (tid == 0) st[0] = mk_h2(1.f, 0.f);
    if (tid < NQ) red[tid] = 0.f;

    // ---- precompute all 64 single-qubit 2x2 gates (complex f32) ----
    if (tid < QDEPTH * NQ) {
        const int d = tid >> 4, q = tid & 15;
        const float a = tanhf(x[b * NQ + q]);
        const float* p = params + (d * NQ + q) * 3;
        const float tx = a + p[0], ty = a + p[1], tz = a + p[2];
        const float cx = cosf(0.5f * tx), sx = sinf(0.5f * tx);
        const float cy = cosf(0.5f * ty), sy = sinf(0.5f * ty);
        const float er = cosf(0.5f * tz), ei = -sinf(0.5f * tz); // e^{-i tz/2}
        // M = RY@RX
        const float m00r =  cy * cx, m00i =  sy * sx;
        const float m01r = -sy * cx, m01i = -cy * sx;
        const float m10r =  sy * cx, m10i = -cy * sx;
        const float m11r =  cy * cx, m11i = -sy * sx;
        float* gm = gates + tid * 8;
        gm[0] = m00r * er - m00i * ei;  gm[1] = m00r * ei + m00i * er;
        gm[2] = m01r * er - m01i * ei;  gm[3] = m01r * ei + m01i * er;
        gm[4] = m10r * er + m10i * ei;  gm[5] = m10i * er - m10r * ei;
        gm[6] = m11r * er + m11i * ei;  gm[7] = m11i * er - m11r * ei;
    }
    __syncthreads();

    #pragma unroll
    for (int d = 0; d < QDEPTH; ++d) {
        #pragma unroll
        for (int grp = 0; grp < 4; ++grp) {
            // Build A_real = [Ur | -Ui], A_imag = [Ui | Ur] (16x32 f16) from
            // U4 = kron of the group's four 2x2 gates; 256 threads, 1 entry each.
            {
                const int r = tid >> 4, c = tid & 15;
                float pr = 1.f, pi = 0.f;
                #pragma unroll
                for (int u = 0; u < 4; ++u) {
                    const int rb = (r >> (3 - u)) & 1;
                    const int cb = (c >> (3 - u)) & 1;
                    const float* e = gates + ((d * NQ + (4 * grp + u)) * 4 + rb * 2 + cb) * 2;
                    const float nr = pr * e[0] - pi * e[1];
                    const float ni = pr * e[1] + pi * e[0];
                    pr = nr; pi = ni;
                }
                Are[r * 32 + c]      = (_Float16)pr;
                Are[r * 32 + 16 + c] = (_Float16)(-pi);
                Aim[r * 32 + c]      = (_Float16)pi;
                Aim[r * 32 + 16 + c] = (_Float16)pr;
            }
            __syncthreads();

            // Load A fragments in contiguous k order: K = 16*(lane/16) + hh.
            v16h ar, ai;
            {
                const int M = lane & 15;
                #pragma unroll
                for (int hh = 0; hh < 16; ++hh) {
                    const int K = hh + ((lane < 16) ? 0 : 16);
                    ar[hh] = Are[M * 32 + K];
                    ai[hh] = Aim[M * 32 + K];
                }
            }

            const int s = 4 * (3 - grp);        // nibble of this group's logical bits
            const int lowmask = (1 << s) - 1;
            // Physical row-offset masks for the group nibble (constants).
            unsigned R[16];
            #pragma unroll
            for (int r = 0; r < 16; ++r) {
                unsigned v = 0;
                #pragma unroll
                for (int j = 0; j < 4; ++j)
                    if ((r >> j) & 1) v ^= kPerm.S[d][s + j];
                R[r] = v;
            }
            const unsigned hib = (lane < 16) ? 0u : R[8];

            // 256 column-tiles of 16 columns; 32 per wave; uniform -> EXEC all 1s.
            for (int tile = wave * 32; tile < wave * 32 + 32; ++tile) {
                const int n = lane & 15;
                const int col = tile * 16 + n;                 // 12-bit column index
                const int lcol = ((col >> s) << (s + 4)) | (col & lowmask);
                unsigned pb = 0;                               // physical base (sigma)
                #pragma unroll
                for (int bb = 0; bb < 16; ++bb)
                    pb ^= (0u - (((unsigned)lcol >> bb) & 1u)) & kPerm.S[d][bb];

                // B fragment: rows 0..15 = Sr (lanes<16), rows 16..31 = Si (lanes>=16)
                v16h bf;
                #pragma unroll
                for (int hh = 0; hh < 16; ++hh) {
                    const h2 v = st[pb ^ R[hh]];
                    bf[hh] = (lane < 16) ? v.x : v.y;
                }
                v8f cz = {};
                v8f cr = __builtin_amdgcn_wmma_f32_16x16x32_f16(
                             false, ar, false, bf, (short)0, cz, false, false);
                v8f ci = __builtin_amdgcn_wmma_f32_16x16x32_f16(
                             false, ai, false, bf, (short)0, cz, false, false);
                // C scatter: VGPR v -> row v (lanes<16) / row 8+v (lanes>=16)
                #pragma unroll
                for (int v = 0; v < 8; ++v) {
                    st[pb ^ R[v] ^ hib] = mk_h2(cr[v], ci[v]);
                }
            }
            __syncthreads();
        }
        // CNOT block of this layer: absorbed into sigma (kPerm) -- no data motion.
    }

    // ===== Measure: Z expectation per qubit (logical walk through sigma_4) =====
    float acc[NQ];
    #pragma unroll
    for (int q = 0; q < NQ; ++q) acc[q] = 0.f;
    for (int l = tid; l < QDIM; l += BLK) {
        unsigned ip = 0;
        #pragma unroll
        for (int bb = 0; bb < 16; ++bb)
            ip ^= (0u - (((unsigned)l >> bb) & 1u)) & kPerm.S[QDEPTH][bb];
        const h2 v = st[ip];
        const float re = (float)v.x, im = (float)v.y;
        const float p = re * re + im * im;
        #pragma unroll
        for (int q = 0; q < NQ; ++q)
            acc[q] += ((l >> (15 - q)) & 1) ? -p : p;
    }
    #pragma unroll
    for (int q = 0; q < NQ; ++q)
        for (int off = 16; off > 0; off >>= 1)
            acc[q] += __shfl_xor(acc[q], off, 32);
    if (lane == 0) {
        #pragma unroll
        for (int q = 0; q < NQ; ++q) atomicAdd(&red[q], acc[q]);
    }
    __syncthreads();
    if (tid < NQ) out[b * NQ + tid] = red[tid];
}

extern "C" void kernel_launch(void* const* d_in, const int* in_sizes, int n_in,
                              void* d_out, int out_size, void* d_ws, size_t ws_size,
                              hipStream_t stream) {
    (void)n_in; (void)d_ws; (void)ws_size; (void)out_size;
    const float* x      = (const float*)d_in[0];   // [B,16] f32
    const float* params = (const float*)d_in[1];   // [4,16,3] f32
    float* out = (float*)d_out;                    // [B,16] f32

    const int batch = in_sizes[0] / NQ;            // 512

    // LDS: state 256KB + gates 2KB + A frags 2KB + reduction 64B
    const size_t smem = (size_t)QDIM * 4
                      + (size_t)QDEPTH * NQ * 8 * sizeof(float)
                      + 1024 * sizeof(_Float16)
                      + NQ * sizeof(float);
    (void)hipFuncSetAttribute((const void*)qsim_kernel,
                              hipFuncAttributeMaxDynamicSharedMemorySize,
                              (int)smem);
    qsim_kernel<<<batch, BLK, smem, stream>>>(x, params, out);
}
